// RecurrentUnitAtt_17755394801910
// MI455X (gfx1250) — matrix operand order repaired and checked by reading, not compiled, not run on hardware
//
#include <hip/hip_runtime.h>

typedef __attribute__((ext_vector_type(16))) _Float16 v16h;
typedef __attribute__((ext_vector_type(8)))  _Float16 v8h;
typedef __attribute__((ext_vector_type(8)))  float    v8f;

#define B_SZ 8
#define NPTS 4096
#define KNN  16
#define CIN  64
#define CPAD 96        // 67 input channels padded to 3 K-steps of 32
#define NW0  192       // 3 gates x 64 outputs
#define TQ   16        // queries per workgroup
#define ROWS 256       // TQ * KNN rows per workgroup

// ---------------------------------------------------------------------------
// Kernel 1: brute-force KNN (16 nearest of 4096) — VALU + LDS tiling
// ---------------------------------------------------------------------------
__global__ void __launch_bounds__(256) knn_kernel(const float* __restrict__ xyz1,
                                                  const float* __restrict__ xyz2,
                                                  int* __restrict__ idx_out) {
  __shared__ float sR[3][256];
  int b  = blockIdx.x >> 4;          // 16 blocks of 256 queries per batch
  int q0 = (blockIdx.x & 15) << 8;
  int t  = threadIdx.x;
  int n  = q0 + t;
  const float* x1b = xyz1 + (size_t)b * 3 * NPTS;
  const float* x2b = xyz2 + (size_t)b * 3 * NPTS;
  float qx = x1b[n], qy = x1b[NPTS + n], qz = x1b[2 * NPTS + n];

  float bd[KNN]; int bi[KNN];
#pragma unroll
  for (int s = 0; s < KNN; ++s) { bd[s] = 3.4e38f; bi[s] = 0; }

  for (int m0 = 0; m0 < NPTS; m0 += 256) {
    __syncthreads();
    sR[0][t] = x2b[m0 + t];
    sR[1][t] = x2b[NPTS + m0 + t];
    sR[2][t] = x2b[2 * NPTS + m0 + t];
    __syncthreads();
    for (int j = 0; j < 256; ++j) {
      float dx = sR[0][j] - qx, dy = sR[1][j] - qy, dz = sR[2][j] - qz;
      float d  = dx * dx + dy * dy + dz * dz;
      if (d < bd[KNN - 1]) {
        int id = m0 + j;
#pragma unroll
        for (int s = 0; s < KNN; ++s) {
          if (d < bd[s]) {
            float td = bd[s]; int ti = bi[s];
            bd[s] = d; bi[s] = id; d = td; id = ti;
          }
        }
      }
    }
  }
  int* o = idx_out + ((size_t)(b * NPTS + n)) * KNN;
#pragma unroll
  for (int s = 0; s < KNN; ++s) o[s] = bi[s];
}

// ---------------------------------------------------------------------------
// Kernel 2: pack f32 weights -> f16, layer0 gates concatenated + K padded
// ---------------------------------------------------------------------------
__global__ void pack_weights(const float* __restrict__ wr0, const float* __restrict__ wz0,
                             const float* __restrict__ wh0, const float* __restrict__ wr1,
                             const float* __restrict__ wro,
                             _Float16* __restrict__ W0h, _Float16* __restrict__ W1h) {
  int i = blockIdx.x * 256 + threadIdx.x;
  if (i < NW0 * CPAD) {
    int o = i / CPAD, c = i % CPAD;
    int g = o >> 6, oo = o & 63;
    float v = 0.f;
    if (c < 67) {
      const float* w = (g == 0) ? wr0 : ((g == 1) ? wz0 : wh0);
      v = w[oo * 67 + c];
    }
    W0h[i] = (_Float16)v;
  }
  int j = i - NW0 * CPAD;
  if (j >= 0 && j < 128 * 64) {
    int o = j >> 6, c = j & 63;
    float v = (o < 64) ? wr1[o * 64 + c] : wro[(o - 64) * 64 + c];
    W1h[j] = (_Float16)v;
  }
}

// ---------------------------------------------------------------------------
// WMMA fragment helpers (wave32 16x16x32 f16 layouts per CDNA5 ISA 7.12.2)
// ---------------------------------------------------------------------------
__device__ __forceinline__ v16h load_a_frag(const _Float16* base, int ld,
                                            int row0, int k0, int lane) {
  int r = lane & 15, hi = lane >> 4;           // A: lane = row; K {0-7,16-23}/{8-15,24-31}
  const _Float16* p = base + (size_t)(row0 + r) * ld + k0 + hi * 8;
  v8h lo = *(const v8h*)p;
  v8h hh = *(const v8h*)(p + 16);
  v16h out;
#pragma unroll
  for (int i = 0; i < 8; ++i) { out[i] = lo[i]; out[i + 8] = hh[i]; }
  return out;
}

__device__ __forceinline__ v16h load_b_frag(const _Float16* base, int ld,
                                            int col0, int k0, int lane) {
  int c = lane & 15, hi = lane >> 4;           // B: lane = col; K {0-7,16-23}/{8-15,24-31}
  const _Float16* p = base + (size_t)(col0 + c) * ld + k0 + hi * 8;
  v8h lo = *(const v8h*)p;
  v8h hh = *(const v8h*)(p + 16);
  v16h out;
#pragma unroll
  for (int i = 0; i < 8; ++i) { out[i] = lo[i]; out[i + 8] = hh[i]; }
  return out;
}

__device__ __forceinline__ v8f wmma16(v16h a, v16h b, v8f c) {
  return __builtin_amdgcn_wmma_f32_16x16x32_f16(false, a, false, b, (short)0, c,
                                                false, false);
}

__device__ __forceinline__ float leaky(float v) { return (v >= 0.f) ? v : 0.1f * v; }

// ---------------------------------------------------------------------------
// Kernel 3: fused gather + 3-gate GRU MLP stack on WMMA
// ---------------------------------------------------------------------------
__global__ void __launch_bounds__(256) gru_fused(
    const float* __restrict__ xyz1, const float* __restrict__ xyz2,
    const float* __restrict__ points1, const float* __restrict__ points2,
    const float* __restrict__ w_fuse_r, const float* __restrict__ w_fuse_z,
    const float* __restrict__ b_r0, const float* __restrict__ b_z0,
    const float* __restrict__ b_h0, const float* __restrict__ b_r1,
    const float* __restrict__ w_z1, const float* __restrict__ b_z1,
    const float* __restrict__ w_h1, const float* __restrict__ b_h1,
    const int* __restrict__ idxbuf,
    const _Float16* __restrict__ W0h, const _Float16* __restrict__ W1h,
    float* __restrict__ out) {
  extern __shared__ __align__(16) char smem[];
  _Float16* sX  = (_Float16*)smem;              // [256][96]  new_points  48KB
  _Float16* sW0 = sX + ROWS * CPAD;             // [192][96]  layer0 W    36KB
  _Float16* sW1 = sW0 + NW0 * CPAD;             // [128][64]  w_r1|w_ro   16KB
  _Float16* sXi = sW1 + 128 * 64;               // [256][64]  interm A    32KB
  float* sH0 = (float*)(sXi + ROWS * 64);       // [256][64]  h pre-act   64KB
  float* sP1 = sH0 + ROWS * 64;                 // [16][64]
  float* sFR = sP1 + TQ * 64;                   // [16][64]
  float* sFZ = sFR + TQ * 64;                   // [16][64]
  float* sZP = sFZ + TQ * 64;                   // [16][64]
  float* sHP = sZP + TQ * 64;                   // [16][64]

  int b  = blockIdx.x >> 8;                     // 256 WGs per batch
  int q0 = (blockIdx.x & 255) * TQ;
  int tid = threadIdx.x;
  int lane = tid & 31, wid = tid >> 5;
  int ncol = lane & 15, hi = lane >> 4;

  const float* p2b = points2 + (size_t)b * CIN * NPTS;
  const float* p1b = points1 + (size_t)b * CIN * NPTS;
  const float* x1b = xyz1 + (size_t)b * 3 * NPTS;
  const float* x2b = xyz2 + (size_t)b * 3 * NPTS;

  // ---- stage weights (vector copies) ----
  {
    const unsigned long long* s0 = (const unsigned long long*)W0h;
    unsigned long long* d0 = (unsigned long long*)sW0;
    for (int i = tid; i < NW0 * CPAD / 4; i += 256) d0[i] = s0[i];
    const unsigned long long* s1 = (const unsigned long long*)W1h;
    unsigned long long* d1 = (unsigned long long*)sW1;
    for (int i = tid; i < 128 * 64 / 4; i += 256) d1[i] = s1[i];
  }
  // ---- stage p1 tile ----
  for (int e = tid; e < TQ * 64; e += 256) {
    int q = e >> 6, c = e & 63;
    sP1[e] = p1b[(size_t)c * NPTS + q0 + q];
  }
  // ---- gather new_points: row = q*16 + k ----
  {
    int r = tid, q = r >> 4, k = r & 15;
    int n = q0 + q;
    int j = idxbuf[((size_t)(b * NPTS + n)) * KNN + k];
    _Float16* xr = sX + r * CPAD;
#pragma unroll 4
    for (int c = 0; c < CIN; ++c) xr[c] = (_Float16)p2b[(size_t)c * NPTS + j];
#pragma unroll
    for (int c = 0; c < 3; ++c)
      xr[CIN + c] = (_Float16)(x2b[c * NPTS + j] - x1b[c * NPTS + n]);
#pragma unroll
    for (int c = 67; c < CPAD; ++c) xr[c] = (_Float16)0.f;
  }
  __syncthreads();

  // ---- fuse_r / fuse_z = p1 x W  (small, VALU) ----
  for (int e = tid; e < TQ * 64; e += 256) {
    int q = e >> 6, o = e & 63;
    float ar = 0.f, az = 0.f;
    const float* p = sP1 + q * 64;
#pragma unroll 8
    for (int c = 0; c < 64; ++c) {
      float pv = p[c];
      ar += pv * w_fuse_r[o * 64 + c];
      az += pv * w_fuse_z[o * 64 + c];
    }
    sFR[e] = ar; sFZ[e] = az;
  }
  __syncthreads();

  // ---- pass 1: [256x96] x [96x192]  (r|z|h), fused epilogues ----
  for (int t = wid; t < TQ * 12; t += 8) {
    int q = t / 12, nt = t % 12;
    int gate = nt >> 2, o0 = (nt & 3) * 16;
    v8f acc = {};
#pragma unroll
    for (int ks = 0; ks < 3; ++ks) {
      v16h a = load_a_frag(sX, CPAD, q * 16, ks * 32, lane);
      v16h w = load_b_frag(sW0, CPAD, gate * 64 + o0, ks * 32, lane);
      acc = wmma16(a, w, acc);
    }
    int o = o0 + ncol;
    if (gate == 0) {                                   // r gate -> sXi (f16 A)
      float bo = b_r0[o], fr = sFR[q * 64 + o];
#pragma unroll
      for (int i = 0; i < 8; ++i) {
        float v = leaky(acc[i] + bo + fr);
        sXi[(q * 16 + i + 8 * hi) * 64 + o] = (_Float16)v;
      }
    } else if (gate == 1) {                            // z gate -> max-pool(k)
      float bo = b_z0[o], fz = sFZ[q * 64 + o];
      float m = -3.4e38f;
#pragma unroll
      for (int i = 0; i < 8; ++i) m = fmaxf(m, leaky(acc[i] + bo + fz));
      m = fmaxf(m, __shfl_xor(m, 16, 32));
      if (hi == 0) sZP[q * 64 + o] = m;
    } else {                                           // h gate pre-act -> LDS
      float bo = b_h0[o];
#pragma unroll
      for (int i = 0; i < 8; ++i)
        sH0[(q * 16 + i + 8 * hi) * 64 + o] = acc[i] + bo;
    }
  }
  __syncthreads();

  // ---- pass 2: r1 GEMM, keep results in regs, then sigmoid*p1 -> sXi ----
  v8f racc[8];
#pragma unroll
  for (int j = 0; j < 8; ++j) {
    int t = wid + j * 8;
    int q = t >> 2, o0 = (t & 3) * 16;
    v8f acc = {};
#pragma unroll
    for (int ks = 0; ks < 2; ++ks) {
      v16h a = load_a_frag(sXi, 64, q * 16, ks * 32, lane);
      v16h w = load_b_frag(sW1, 64, o0, ks * 32, lane);
      acc = wmma16(a, w, acc);
    }
    racc[j] = acc;
  }
  __syncthreads();                 // everyone done reading sXi
#pragma unroll
  for (int j = 0; j < 8; ++j) {
    int t = wid + j * 8;
    int q = t >> 2, o0 = (t & 3) * 16;
    int o = o0 + ncol;
    float bo = b_r1[o];
#pragma unroll
    for (int i = 0; i < 8; ++i) {
      float x = racc[j][i] + bo;
      float s = 1.f / (1.f + __expf(-x));
      sXi[(q * 16 + i + 8 * hi) * 64 + o] = (_Float16)(s * sP1[q * 64 + o]);
    }
  }
  __syncthreads();

  // ---- pass 3: pe GEMM (x w_fuse_ro) + h activation + max-pool(k) ----
#pragma unroll
  for (int j = 0; j < 8; ++j) {
    int t = wid + j * 8;
    int q = t >> 2, o0 = (t & 3) * 16;
    v8f acc = {};
#pragma unroll
    for (int ks = 0; ks < 2; ++ks) {
      v16h a = load_a_frag(sXi, 64, q * 16, ks * 32, lane);
      v16h w = load_b_frag(sW1, 64, 64 + o0, ks * 32, lane);
      acc = wmma16(a, w, acc);
    }
    int o = o0 + ncol;
    float m = -3.4e38f;
#pragma unroll
    for (int i = 0; i < 8; ++i) {
      float v = leaky(sH0[(q * 16 + i + 8 * hi) * 64 + o] + acc[i]);
      m = fmaxf(m, v);
    }
    m = fmaxf(m, __shfl_xor(m, 16, 32));
    if (hi == 0) sHP[q * 64 + o] = m;
  }
  __syncthreads();

  // ---- final: z1 / h1 (16 rows, VALU) + GRU blend ----
  for (int e = tid; e < TQ * 64; e += 256) {
    int q = e >> 6, c = e & 63;
    float za = b_z1[c], ha = b_h1[c];
#pragma unroll 8
    for (int cc = 0; cc < 64; ++cc) {
      za += sZP[q * 64 + cc] * w_z1[c * 64 + cc];
      ha += sHP[q * 64 + cc] * w_h1[c * 64 + cc];
    }
    float z = 1.f / (1.f + __expf(-za));
    float h = tanhf(ha);
    float p = sP1[q * 64 + c];
    out[((size_t)b * CIN + c) * NPTS + q0 + q] = (1.f - z) * p + z * h;
  }
}

// ---------------------------------------------------------------------------
extern "C" void kernel_launch(void* const* d_in, const int* in_sizes, int n_in,
                              void* d_out, int out_size, void* d_ws, size_t ws_size,
                              hipStream_t stream) {
  (void)in_sizes; (void)n_in; (void)out_size; (void)ws_size;
  const float* xyz1      = (const float*)d_in[0];
  const float* xyz2      = (const float*)d_in[1];
  const float* points1   = (const float*)d_in[2];
  const float* points2   = (const float*)d_in[3];
  const float* w_fuse_r  = (const float*)d_in[4];
  const float* w_fuse_z  = (const float*)d_in[5];
  const float* w_fuse_ro = (const float*)d_in[6];
  const float* w_r0 = (const float*)d_in[7];
  const float* b_r0 = (const float*)d_in[8];
  const float* w_r1 = (const float*)d_in[9];
  const float* b_r1 = (const float*)d_in[10];
  const float* w_z0 = (const float*)d_in[11];
  const float* b_z0 = (const float*)d_in[12];
  const float* w_z1 = (const float*)d_in[13];
  const float* b_z1 = (const float*)d_in[14];
  const float* w_h0 = (const float*)d_in[15];
  const float* b_h0 = (const float*)d_in[16];
  const float* w_h1 = (const float*)d_in[17];
  const float* b_h1 = (const float*)d_in[18];
  float* out = (float*)d_out;

  char* ws = (char*)d_ws;
  int* idxbuf = (int*)ws;
  size_t off = (size_t)B_SZ * NPTS * KNN * sizeof(int);   // 2 MB
  _Float16* W0h = (_Float16*)(ws + off);
  _Float16* W1h = W0h + NW0 * CPAD;

  knn_kernel<<<B_SZ * NPTS / 256, 256, 0, stream>>>(xyz1, xyz2, idxbuf);
  pack_weights<<<(NW0 * CPAD + 128 * 64 + 255) / 256, 256, 0, stream>>>(
      w_r0, w_z0, w_h0, w_r1, w_fuse_ro, W0h, W1h);

  size_t smem = (size_t)ROWS * CPAD * 2 + NW0 * CPAD * 2 + 128 * 64 * 2 +
                ROWS * 64 * 2 + ROWS * 64 * 4 + TQ * 64 * 4 * 5;   // 221184 B
  gru_fused<<<B_SZ * (NPTS / TQ), 256, smem, stream>>>(
      xyz1, xyz2, points1, points2, w_fuse_r, w_fuse_z,
      b_r0, b_z0, b_h0, b_r1, w_z1, b_z1, w_h1, b_h1,
      idxbuf, W0h, W1h, out);
}
